// GNNEncoder_39642548142225
// MI455X (gfx1250) — matrix operand order repaired and checked by reading, not compiled, not run on hardware
//
#include <hip/hip_runtime.h>
#include <hip/hip_bf16.h>
#include <stdint.h>

// ---------------------------------------------------------------------------
// GAT 2-layer encoder for MI455X (gfx1250, wave32).
// Dense projections run on v_wmma_f32_16x16x32_bf16 with a bf16 hi/lo split
// (3 WMMA products) for ~fp32 accuracy. Edge softmax/aggregation uses f32
// atomics; all hot buffers (h_src 12.8MB, agg 12.8MB) are L2-resident.
// ---------------------------------------------------------------------------

#define NEG_SLOPE 0.2f

typedef __attribute__((ext_vector_type(16))) __bf16 v16bf;
typedef __attribute__((ext_vector_type(8)))  float  v8f;

// ---- fp32 -> bf16 (round-to-nearest-even) helpers -------------------------
__device__ __forceinline__ uint32_t bf16_rne(float f) {
  uint32_t u = __float_as_uint(f);
  return ((u + 0x7FFFu + ((u >> 16) & 1u)) >> 16) & 0xFFFFu;
}
__device__ __forceinline__ float bf16_to_f32(uint32_t b) {
  return __uint_as_float(b << 16);
}

// ---- monotone float<->uint key for atomicMax on floats --------------------
__device__ __forceinline__ uint32_t fkey(float f) {
  uint32_t u = __float_as_uint(f);
  return (u & 0x80000000u) ? ~u : (u | 0x80000000u);
}
__device__ __forceinline__ float funkey(uint32_t k) {
  return (k & 0x80000000u) ? __uint_as_float(k & 0x7FFFFFFFu)
                           : __uint_as_float(~k);
}

__device__ __forceinline__ float lrelu(float x) {
  return (x > 0.f) ? x : NEG_SLOPE * x;
}

// ---------------------------------------------------------------------------
// Pack X [N,K] fp32 (row-major) into per-lane WMMA A-fragment layout, split
// into bf16 hi + lo planes.  16-bit A 16x32 layout (ISA 7.12.2):
//   lane L: row M = L%16, half = L/16
//   VGPR i (0..3): K = kbase + 8*half + 2i,2i+1
//   VGPR i (4..7): K = kbase + 8*half + 16 + 2(i-4), +1
// Flat dword index: ((rowTile*nKStep + s)*32 + L)*8 + i
// ---------------------------------------------------------------------------
__global__ void pack_x_kernel(const float* __restrict__ x,
                              uint32_t* __restrict__ ph,
                              uint32_t* __restrict__ pl,
                              int N, int K) {
  const int nKStep = K >> 5;
  long o = (long)blockIdx.x * blockDim.x + threadIdx.x;
  long total = (long)((N + 15) >> 4) * nKStep * 256;
  if (o >= total) return;
  int  i    = (int)(o & 7);
  int  L    = (int)((o >> 3) & 31);
  int  s    = (int)((o >> 8) % nKStep);
  long t    = o / (256L * nKStep);
  int  row  = (int)t * 16 + (L & 15);
  int  half = L >> 4;
  int  k    = s * 32 + 8 * half + ((i < 4) ? (2 * i) : (16 + 2 * (i - 4)));
  float x0 = 0.f, x1 = 0.f;
  if (row < N) {
    x0 = x[(size_t)row * K + k];
    x1 = x[(size_t)row * K + k + 1];
  }
  uint32_t h0 = bf16_rne(x0), h1 = bf16_rne(x1);
  float    r0 = x0 - bf16_to_f32(h0), r1 = x1 - bf16_to_f32(h1);
  uint32_t l0 = bf16_rne(r0), l1 = bf16_rne(r1);
  ph[o] = h0 | (h1 << 16);
  pl[o] = l0 | (l1 << 16);
}

// ---------------------------------------------------------------------------
// Pack W [K,64] fp32 into per-lane WMMA B-fragment layout (4 column tiles),
// bf16 hi/lo.  16-bit B 32x16: lane L: col N = cTile*16 + L%16, half = L/16,
// VGPR i (0..3): K = kbase + 8*half + 2i, 2i+1; VGPR i (4..7): K = kbase + 8*half + 16 + 2(i-4), +1.
// Flat dword index: ((cTile*nKStep + s)*32 + L)*8 + i
// ---------------------------------------------------------------------------
__global__ void pack_w_kernel(const float* __restrict__ W,
                              uint32_t* __restrict__ ph,
                              uint32_t* __restrict__ pl,
                              int K) {
  const int nKStep = K >> 5;
  int o = blockIdx.x * blockDim.x + threadIdx.x;
  int total = 4 * nKStep * 256;
  if (o >= total) return;
  int i    = o & 7;
  int L    = (o >> 3) & 31;
  int s    = (o >> 8) % nKStep;
  int cT   = o / (256 * nKStep);
  int col  = cT * 16 + (L & 15);
  int half = L >> 4;
  int k    = s * 32 + 8 * half + ((i < 4) ? (2 * i) : (16 + 2 * (i - 4)));
  float w0 = W[(size_t)k * 64 + col];
  float w1 = W[(size_t)(k + 1) * 64 + col];
  uint32_t h0 = bf16_rne(w0), h1 = bf16_rne(w1);
  float    r0 = w0 - bf16_to_f32(h0), r1 = w1 - bf16_to_f32(h1);
  uint32_t l0 = bf16_rne(r0), l1 = bf16_rne(r1);
  ph[o] = h0 | (h1 << 16);
  pl[o] = l0 | (l1 << 16);
}

// ---------------------------------------------------------------------------
// v_dst[k] = sum_h W_dst[k,h] * a_dst[h]   (folds h_dst @ a_dst into a GEMV)
// ---------------------------------------------------------------------------
__global__ void vdst_kernel(const float* __restrict__ Wd,
                            const float* __restrict__ adst,
                            float* __restrict__ v, int K) {
  int k = threadIdx.x;
  if (k >= K) return;
  float s = 0.f;
  for (int h = 0; h < 64; ++h) s += Wd[(size_t)k * 64 + h] * adst[h];
  v[k] = s;
}

__device__ __forceinline__ v16bf load_frag(const uint32_t* p) {
  union { uint4 q[2]; v16bf v; } f;
  f.q[0] = *reinterpret_cast<const uint4*>(p);
  f.q[1] = *reinterpret_cast<const uint4*>(p + 4);
  return f.v;
}

// ---------------------------------------------------------------------------
// h[N,64] = X[N,K] @ W[K,64] with bf16 hi/lo split:
//   C += Ahi*Bhi + Ahi*Blo + Alo*Bhi   (drops the ~2^-16 Alo*Blo term)
// Block: 128 threads = 4 waves; wave w owns column tile [16w,16w+16).
// Block b owns row tile [16b,16b+16).  3 v_wmma per 32-wide K step.
// ---------------------------------------------------------------------------
__global__ void gat_gemm_wmma(const uint32_t* __restrict__ ah_,
                              const uint32_t* __restrict__ al_,
                              const uint32_t* __restrict__ bh_,
                              const uint32_t* __restrict__ bl_,
                              float* __restrict__ out, int N, int nKStep) {
  const int lane = threadIdx.x & 31;
  const int wv   = threadIdx.x >> 5;   // column tile 0..3
  const int rt   = blockIdx.x;         // row tile
  v8f c = {0.f, 0.f, 0.f, 0.f, 0.f, 0.f, 0.f, 0.f};
  for (int s = 0; s < nKStep; ++s) {
    size_t ao = (((size_t)rt * nKStep + s) * 32 + lane) * 8;
    size_t bo = (((size_t)wv * nKStep + s) * 32 + lane) * 8;
    v16bf ah = load_frag(ah_ + ao);
    v16bf al = load_frag(al_ + ao);
    v16bf bh = load_frag(bh_ + bo);
    v16bf bl = load_frag(bl_ + bo);
    c = __builtin_amdgcn_wmma_f32_16x16x32_bf16(false, ah, false, bh,
                                                (short)0, c, false, false);
    c = __builtin_amdgcn_wmma_f32_16x16x32_bf16(false, ah, false, bl,
                                                (short)0, c, false, false);
    c = __builtin_amdgcn_wmma_f32_16x16x32_bf16(false, al, false, bh,
                                                (short)0, c, false, false);
  }
  // C/D 16x16 f32 layout: VGPR j, lane L -> M = j + 8*(L/16), N = L%16
  const int half = lane >> 4, nn = lane & 15;
#pragma unroll
  for (int j = 0; j < 8; ++j) {
    int row = rt * 16 + j + 8 * half;
    if (row < N) out[(size_t)row * 64 + wv * 16 + nn] = c[j];
  }
}

// ---------------------------------------------------------------------------
// One wave per node: alpha_src[n] = h[n,:] . a_src ; alpha_dst[n] = x[n,:] . v
// ---------------------------------------------------------------------------
__global__ void alpha_kernel(const float* __restrict__ h,
                             const float* __restrict__ asrc,
                             const float* __restrict__ xin,
                             const float* __restrict__ vd,
                             float* __restrict__ as, float* __restrict__ ad,
                             int N, int K) {
  int wid  = (int)((blockIdx.x * (long)blockDim.x + threadIdx.x) >> 5);
  int lane = threadIdx.x & 31;
  if (wid >= N) return;
  float s1 = h[(size_t)wid * 64 + lane] * asrc[lane] +
             h[(size_t)wid * 64 + lane + 32] * asrc[lane + 32];
  float s2 = 0.f;
  for (int k = lane; k < K; k += 32) s2 += xin[(size_t)wid * K + k] * vd[k];
  for (int m = 16; m; m >>= 1) {
    s1 += __shfl_xor(s1, m, 32);
    s2 += __shfl_xor(s2, m, 32);
  }
  if (lane == 0) { as[wid] = s1; ad[wid] = s2; }
}

// ---- edge pass 1: segment max of leaky-relu logits (keyed atomicMax) ------
__global__ void edge_max_kernel(const int* __restrict__ src,
                                const int* __restrict__ dst,
                                const float* __restrict__ as,
                                const float* __restrict__ ad,
                                uint32_t* __restrict__ mu, int E) {
  int e = blockIdx.x * blockDim.x + threadIdx.x;
  if (e >= E) return;
  float l = lrelu(as[src[e]] + ad[dst[e]]);
  atomicMax(&mu[dst[e]], fkey(l));
}

// ---- edge pass 2: denom[dst] += exp(logit - max) --------------------------
__global__ void edge_sum_kernel(const int* __restrict__ src,
                                const int* __restrict__ dst,
                                const float* __restrict__ as,
                                const float* __restrict__ ad,
                                const uint32_t* __restrict__ mu,
                                float* __restrict__ den, int E) {
  int e = blockIdx.x * blockDim.x + threadIdx.x;
  if (e >= E) return;
  int d = dst[e];
  float l = lrelu(as[src[e]] + ad[d]);
  atomicAdd(&den[d], expf(l - funkey(mu[d])));
}

// ---- edge pass 3: agg[dst,h] += h_src[src,h] * softmax weight -------------
// One thread per (edge, channel); each wave covers 32 channels of one edge.
__global__ void edge_agg_kernel(const int* __restrict__ src,
                                const int* __restrict__ dst,
                                const float* __restrict__ as,
                                const float* __restrict__ ad,
                                const uint32_t* __restrict__ mu,
                                const float* __restrict__ den,
                                const float* __restrict__ h,
                                float* __restrict__ agg, long total) {
  long t = blockIdx.x * (long)blockDim.x + threadIdx.x;
  if (t >= total) return;
  int e  = (int)(t >> 6);
  int hh = (int)(t & 63);
  int s  = src[e], d = dst[e];
  float l  = lrelu(as[s] + ad[d]);
  float wgt = expf(l - funkey(mu[d])) / den[d];
  atomicAdd(&agg[(size_t)d * 64 + hh], h[(size_t)s * 64 + hh] * wgt);
}

// ---- epilogue: out = tanh(agg + bias) -------------------------------------
__global__ void bias_tanh_kernel(const float* __restrict__ agg,
                                 const float* __restrict__ b,
                                 float* __restrict__ out, long total) {
  long t = blockIdx.x * (long)blockDim.x + threadIdx.x;
  if (t >= total) return;
  out[t] = tanhf(agg[t] + b[t & 63]);
}

// ---------------------------------------------------------------------------
extern "C" void kernel_launch(void* const* d_in, const int* in_sizes, int n_in,
                              void* d_out, int out_size, void* d_ws, size_t ws_size,
                              hipStream_t stream) {
  const float* x   = (const float*)d_in[0];
  const int*   src = (const int*)d_in[1];
  const int*   dst = (const int*)d_in[2];
  const float* W1s = (const float*)d_in[3];
  const float* W1d = (const float*)d_in[4];
  const float* a1s = (const float*)d_in[5];
  const float* a1d = (const float*)d_in[6];
  const float* b1  = (const float*)d_in[7];
  const float* W2s = (const float*)d_in[8];
  const float* W2d = (const float*)d_in[9];
  const float* a2s = (const float*)d_in[10];
  const float* a2d = (const float*)d_in[11];
  const float* b2  = (const float*)d_in[12];

  const int D_IN = 128, H = 64;
  const int N = in_sizes[0] / D_IN;
  const int E = in_sizes[1];
  const int nTiles = (N + 15) >> 4;

  // ---- workspace layout (all 256B-aligned, ~65 MB total) ----
  char* w = (char*)d_ws;
  size_t off = 0;
  auto alloc = [&](size_t bytes) -> void* {
    void* p = w + off;
    off += (bytes + 255) & ~(size_t)255;
    return p;
  };
  uint32_t* xph = (uint32_t*)alloc((size_t)nTiles * (D_IN / 32) * 256 * 4);
  uint32_t* xpl = (uint32_t*)alloc((size_t)nTiles * (D_IN / 32) * 256 * 4);
  uint32_t* wph = (uint32_t*)alloc((size_t)4 * (D_IN / 32) * 256 * 4);
  uint32_t* wpl = (uint32_t*)alloc((size_t)4 * (D_IN / 32) * 256 * 4);
  float*    hbuf = (float*)alloc((size_t)N * H * 4);
  float*    agg  = (float*)alloc((size_t)N * H * 4);
  float*    h1   = (float*)alloc((size_t)N * H * 4);
  float*    as   = (float*)alloc((size_t)N * 4);
  float*    ad   = (float*)alloc((size_t)N * 4);
  uint32_t* mu   = (uint32_t*)alloc((size_t)N * 4);
  float*    den  = (float*)alloc((size_t)N * 4);
  float*    vd   = (float*)alloc(128 * 4);
  (void)ws_size; (void)n_in; (void)out_size;

  auto run_layer = [&](const float* xin, int K, const float* Ws, const float* Wd,
                       const float* asrc, const float* adst, const float* bias,
                       float* outbuf) {
    const int nKStep = K / 32;
    hipMemsetAsync(agg, 0, (size_t)N * H * 4, stream);
    hipMemsetAsync(mu,  0, (size_t)N * 4, stream);
    hipMemsetAsync(den, 0, (size_t)N * 4, stream);

    long tpx = (long)nTiles * nKStep * 256;
    pack_x_kernel<<<(unsigned)((tpx + 255) / 256), 256, 0, stream>>>(
        xin, xph, xpl, N, K);
    int tpw = 4 * nKStep * 256;
    pack_w_kernel<<<(unsigned)((tpw + 255) / 256), 256, 0, stream>>>(
        Ws, wph, wpl, K);
    vdst_kernel<<<1, 128, 0, stream>>>(Wd, adst, vd, K);

    gat_gemm_wmma<<<(unsigned)nTiles, 128, 0, stream>>>(
        xph, xpl, wph, wpl, hbuf, N, nKStep);

    alpha_kernel<<<(unsigned)((N + 7) / 8), 256, 0, stream>>>(
        hbuf, asrc, xin, vd, as, ad, N, K);

    edge_max_kernel<<<(unsigned)((E + 255) / 256), 256, 0, stream>>>(
        src, dst, as, ad, mu, E);
    edge_sum_kernel<<<(unsigned)((E + 255) / 256), 256, 0, stream>>>(
        src, dst, as, ad, mu, den, E);
    long tagg = (long)E * H;
    edge_agg_kernel<<<(unsigned)((tagg + 255) / 256), 256, 0, stream>>>(
        src, dst, as, ad, mu, den, hbuf, agg, tagg);

    long tn = (long)N * H;
    bias_tanh_kernel<<<(unsigned)((tn + 255) / 256), 256, 0, stream>>>(
        agg, bias, outbuf, tn);
  };

  run_layer(x,  128, W1s, W1d, a1s, a1d, b1, h1);
  run_layer(h1,  64, W2s, W2d, a2s, a2d, b2, (float*)d_out);
}
